// HybridJediBlock_80590766342681
// MI455X (gfx1250) — matrix operand order repaired and checked by reading, not compiled, not run on hardware
//
#include <hip/hip_runtime.h>
#include <cstdint>

// ---------------- constants (match reference) ----------------
#define BB   2
#define LL   1024
#define DMD  1024
#define NH   16
#define NKV  4
#define HDI  64
#define NST  64
#define NEXP 4
#define FFD  4096
#define TT   (BB*LL)          // 2048 tokens

typedef __bf16 bf16t;
typedef bf16t v16bf __attribute__((ext_vector_type(16)));
typedef float v8f_t __attribute__((ext_vector_type(8)));

__device__ __forceinline__ bf16t f2bf(float f) {
  unsigned int u = __builtin_bit_cast(unsigned int, f);
  unsigned int r = (u + 0x7FFFu + ((u >> 16) & 1u)) >> 16;
  unsigned short s = (unsigned short)r;
  return __builtin_bit_cast(bf16t, s);
}
__device__ __forceinline__ float sigf(float x) { return 1.f / (1.f + __expf(-x)); }
__device__ __forceinline__ float siluf(float x) { return x * sigf(x); }

// load a contiguous 32B LDS run as one WMMA fragment
__device__ __forceinline__ v16bf ld_frag(const unsigned short* p) {
  struct P2 { uint4 x, y; } t;
  t.x = *(const uint4*)p;
  t.y = *(const uint4*)(p + 8);
  return __builtin_bit_cast(v16bf, t);
}

// =====================================================================
// Generic batched bf16 GEMM, f32 accumulate, v_wmma_f32_16x16x32_bf16.
// A: row-major (M x K).  B: N-major (N x K) -- i.e. pre-transposed weights.
// LDS layouts give each lane contiguous 32B fragments (2x ds_load_b128).
// Requires M % BM == 0, N % BN == 0, K % 32 == 0 (true at every call site).
// =====================================================================
struct GemmP {
  const bf16t* A; const bf16t* Bm; float* C;
  long long lda, ldb, ldc;
  int M, N, K;
  int batch, hdiv, bDiv;
  long long aO, aI, bO, bI, cO, cI;
  float alpha;
  const float* bias;
  const float* rowscale; int rsStride;
  int accumulate; int act;   // act: 0 none, 1 sigmoid
};

template<int BWM, int BWN, int WMT, int WNT>
__global__ __launch_bounds__(256) void gemm_bf16(GemmP p) {
  constexpr int BM = BWM * WMT * 16;
  constexpr int BN = BWN * WNT * 16;
  __shared__ unsigned short As[BM][40];   // natural K order: frag = contiguous 16 halfs
  __shared__ unsigned short Bt[BN][40];   // natural K order per n row

  const int z = blockIdx.z;
  const long long offA = (long long)(z / p.hdiv) * p.aO + (long long)(z % p.hdiv) * p.aI;
  const long long offB = (long long)(z / p.hdiv) * p.bO + (long long)((z % p.hdiv) / p.bDiv) * p.bI;
  const long long offC = (long long)(z / p.hdiv) * p.cO + (long long)(z % p.hdiv) * p.cI;
  const unsigned short* __restrict__ Ag = (const unsigned short*)p.A + offA;
  const unsigned short* __restrict__ Bg = (const unsigned short*)p.Bm + offB;
  float* __restrict__ Cg = p.C + offC;

  const int m0 = blockIdx.y * BM;
  const int n0 = blockIdx.x * BN;
  const int tid = threadIdx.x;
  const int wave = tid >> 5, lane = tid & 31;
  const int wm = wave / BWN, wn = wave % BWN;
  const int wMl = wm * WMT * 16, wNl = wn * WNT * 16;
  const int mloc = lane & 15;     // A/B fragment row within 16
  const int khalf = lane >> 4;    // which 16-half K run

  v8f_t c[WMT][WNT] = {};

  for (int k0 = 0; k0 < p.K; k0 += 32) {
    // ---- stage A tile BMx32 (natural K order) ----
    #pragma unroll
    for (int i = 0; i < BM / 32; ++i) {
      int chunk = tid + i * 256;
      int r = chunk >> 3;
      int cc = (chunk & 7) * 4;
      const unsigned short* src = Ag + (long long)(m0 + r) * p.lda + (k0 + cc);
      ushort4 v = *(const ushort4*)src;
      int g = cc >> 3;
      int dst = cc;  // natural K order
      *(ushort4*)&As[r][dst] = v;
      if (i == 0 && k0 + 32 < p.K) __builtin_prefetch(src + 32, 0, 1);
    }
    // ---- stage B tile BNx32 (N-major source, natural K order) ----
    #pragma unroll
    for (int i = 0; i < BN / 32; ++i) {
      int chunk = tid + i * 256;
      int n = chunk >> 3;
      int kc = (chunk & 7) * 4;
      const unsigned short* src = Bg + (long long)(n0 + n) * p.ldb + (k0 + kc);
      ushort4 v = *(const ushort4*)src;
      *(ushort4*)&Bt[n][kc] = v;
      if (i == 0 && k0 + 32 < p.K) __builtin_prefetch(src + 32, 0, 1);
    }
    __syncthreads();

    v16bf af[WMT], bfv[WNT];
    #pragma unroll
    for (int mi = 0; mi < WMT; ++mi)
      af[mi] = ld_frag(&As[wMl + mi * 16 + mloc][khalf * 16]);
    #pragma unroll
    for (int ni = 0; ni < WNT; ++ni)
      bfv[ni] = ld_frag(&Bt[wNl + ni * 16 + mloc][khalf * 16]);
    #pragma unroll
    for (int mi = 0; mi < WMT; ++mi)
      #pragma unroll
      for (int ni = 0; ni < WNT; ++ni)
        c[mi][ni] = __builtin_amdgcn_wmma_f32_16x16x32_bf16(
            false, af[mi], false, bfv[ni], (short)0, c[mi][ni], false, false);
    __syncthreads();
  }

  // ---- epilogue (C/D: lanes 0-15 N=lane, M=r; lanes 16-31 M=r+8) ----
  const int nn = lane & 15, mh = lane >> 4;
  #pragma unroll
  for (int mi = 0; mi < WMT; ++mi)
    #pragma unroll
    for (int ni = 0; ni < WNT; ++ni)
      #pragma unroll
      for (int r = 0; r < 8; ++r) {
        int m = m0 + wMl + mi * 16 + r + 8 * mh;
        int n = n0 + wNl + ni * 16 + nn;
        float v = p.alpha * c[mi][ni][r];
        if (p.bias) v += p.bias[n];
        if (p.act == 1) v = sigf(v);
        if (p.rowscale) v *= p.rowscale[(long long)m * p.rsStride];
        float* cp = Cg + (long long)m * p.ldc + n;
        if (p.accumulate) *cp += v; else *cp = v;
      }
}

// =====================================================================
// Small kernels
// =====================================================================
__global__ __launch_bounds__(256) void k_f2b(const float* __restrict__ s,
                                             bf16t* __restrict__ d, long long n) {
  long long i = (long long)blockIdx.x * 256 + threadIdx.x;
  if (i < n) d[i] = f2bf(s[i]);
}

// f32 (K x N) -> bf16 transposed (N x K), batched on grid.y
__global__ __launch_bounds__(256)
void k_f2b_t(const float* __restrict__ s, bf16t* __restrict__ d, int K, int N) {
  long long base = (long long)blockIdx.y * K * N;
  long long id = (long long)blockIdx.x * 256 + threadIdx.x;
  if (id >= (long long)K * N) return;
  int k = (int)(id / N), n = (int)(id % N);
  d[base + (long long)n * K + k] = f2bf(s[base + id]);
}

// RMSNorm1 + goal_cond + router softmax*mix (one block = one token)
__global__ __launch_bounds__(256)
void k_rms1(const float* __restrict__ x, const float* __restrict__ goal,
            const float* __restrict__ n1w, const float* __restrict__ rw,
            float* __restrict__ xn, bf16t* __restrict__ xnb,
            float* __restrict__ wout) {
  __shared__ float sred[256];
  __shared__ float sr[4];
  const int t = blockIdx.x, tid = threadIdx.x;
  const long long base = (long long)t * DMD;
  float xv[4]; float ss = 0.f;
  #pragma unroll
  for (int i = 0; i < 4; ++i) { xv[i] = x[base + tid * 4 + i]; ss += xv[i] * xv[i]; }
  sred[tid] = ss; __syncthreads();
  for (int s = 128; s > 0; s >>= 1) { if (tid < s) sred[tid] += sred[tid + s]; __syncthreads(); }
  const float rs = rsqrtf(sred[0] / (float)DMD + 1e-6f);
  __syncthreads();
  if (tid < 4) sr[tid] = 0.f;
  __syncthreads();
  float xnv[4];
  #pragma unroll
  for (int i = 0; i < 4; ++i) {
    int cidx = tid * 4 + i;
    xnv[i] = xv[i] * rs * n1w[cidx] + goal[base + cidx];
    xn[base + cidx] = xnv[i];
    xnb[base + cidx] = f2bf(xnv[i]);
  }
  #pragma unroll
  for (int e = 0; e < 4; ++e) {
    float pp = 0.f;
    #pragma unroll
    for (int i = 0; i < 4; ++i) pp += xnv[i] * rw[(tid * 4 + i) * 4 + e];
    atomicAdd(&sr[e], pp);
  }
  __syncthreads();
  if (tid == 0) {
    const float mix[4] = {0.5f, 0.2f, 0.15f, 0.15f};
    float mx = fmaxf(fmaxf(sr[0], sr[1]), fmaxf(sr[2], sr[3]));
    float p[4]; float Z = 0.f;
    #pragma unroll
    for (int e = 0; e < 4; ++e) { p[e] = __expf(sr[e] - mx) * mix[e]; Z += p[e]; }
    #pragma unroll
    for (int e = 0; e < 4; ++e) wout[t * 4 + e] = p[e] / Z;
  }
}

// xn @ xproj_w  (144 cols) per token
__global__ __launch_bounds__(256)
void k_ssmproj(const float* __restrict__ xn, const float* __restrict__ xp,
               float* __restrict__ sp) {
  __shared__ float xs[DMD];
  const int t = blockIdx.x, tid = threadIdx.x;
  for (int i = tid; i < DMD; i += 256) xs[i] = xn[(long long)t * DMD + i];
  __syncthreads();
  if (tid < 144) {
    float acc = 0.f;
    for (int k = 0; k < DMD; ++k) acc += xs[k] * xp[k * 144 + tid];
    sp[(long long)t * 144 + tid] = acc;
  }
}

// Sequential SSD scan: one thread per (b,h,n) chain
__global__ __launch_bounds__(256)
void k_scan(const float* __restrict__ sp, const float* __restrict__ Aw,
            float* __restrict__ states) {
  const int id = blockIdx.x * 256 + threadIdx.x;   // (b*NH+h)*NST+n
  if (id >= BB * NH * NST) return;
  const int n = id & 63, h = (id >> 6) & 15, b = id >> 10;
  const float Av = Aw[h * NST + n];
  float s = 0.f;
  for (int t = 0; t < LL; ++t) {
    const long long tok = (long long)b * LL + t;
    const float xdb = sp[tok * 144 + 128 + h] + sp[tok * 144 + n];
    const float dt = (xdb > 20.f) ? xdb : log1pf(__expf(xdb));
    s = __expf(dt * Av) * s + dt * dt;
    states[(tok * NH + h) * NST + n] = s;
  }
}

// y_h = states . Cs  (one thread per (token, h))
__global__ __launch_bounds__(256)
void k_yh(const float* __restrict__ states, const float* __restrict__ sp,
          float* __restrict__ yh) {
  const int id = blockIdx.x * 256 + threadIdx.x;
  if (id >= TT * NH) return;
  const long long tok = id >> 4; const int h = id & 15;
  float acc = 0.f;
  for (int n = 0; n < NST; ++n)
    acc += states[(tok * NH + h) * NST + n] * sp[tok * 144 + 64 + n];
  yh[id] = acc;
}

// t1 = (y_h + xn*Dp[h]) * gate  -> bf16
__global__ __launch_bounds__(256)
void k_t1(const float* __restrict__ yh, const float* __restrict__ xn,
          const float* __restrict__ Dp, const float* __restrict__ gate,
          bf16t* __restrict__ t1) {
  const long long id = (long long)blockIdx.x * 256 + threadIdx.x;
  if (id >= (long long)TT * DMD) return;
  const int cidx = (int)(id % DMD); const long long tok = id / DMD;
  const int h = cidx >> 6;
  const float v = (yh[tok * NH + h] + xn[id] * Dp[h]) * gate[id];
  t1[id] = f2bf(v);
}

// Head RMS + RoPE, write (b,heads,l,hd) bf16
__global__ __launch_bounds__(64)
void k_rope(const float* __restrict__ src, const float* __restrict__ nw,
            bf16t* __restrict__ dst, int nheads) {
  __shared__ float sv[64]; __shared__ float sred[64];
  const int blk = blockIdx.x;
  const int l = blk % LL;
  const int h = (blk / LL) % nheads;
  const int b = blk / (LL * nheads);
  const int hd = threadIdx.x;
  const float v = src[((long long)(b * LL + l)) * (nheads * HDI) + h * HDI + hd];
  sred[hd] = v * v; __syncthreads();
  for (int s = 32; s > 0; s >>= 1) { if (hd < s) sred[hd] += sred[hd + s]; __syncthreads(); }
  const float rs = rsqrtf(sred[0] / (float)HDI + 1e-6f);
  const float qn = v * rs * nw[hd];
  sv[hd] = qn; __syncthreads();
  const int i2 = hd & 31;
  const float inv = __powf(10000.f, -(float)(2 * i2) / (float)HDI);
  const float ang = (float)l * inv;
  const float rot = (hd < 32) ? -sv[hd + 32] : sv[hd - 32];
  const float o = qn * __cosf(ang) + rot * __sinf(ang);
  dst[(((long long)(b * nheads + h)) * LL + l) * HDI + hd] = f2bf(o);
}

// v (b,l,kvh,hd) f32 -> transposed (b,kvh,hd,l) bf16 (N-major for PV GEMM)
__global__ __launch_bounds__(256)
void k_vperm(const float* __restrict__ v, bf16t* __restrict__ d) {
  const int id = blockIdx.x * 256 + threadIdx.x;
  if (id >= BB * NKV * HDI * LL) return;
  const int l = id & 1023;
  const int hd = (id >> 10) & 63;
  const int kv = (id >> 16) & 3;
  const int b = id >> 18;
  d[id] = f2bf(v[((long long)(b * LL + l)) * (NKV * HDI) + kv * HDI + hd]);
}

// row softmax (1024 cols) f32 -> bf16
__global__ __launch_bounds__(256)
void k_softmax(const float* __restrict__ sc, bf16t* __restrict__ pr) {
  __shared__ float sred[256];
  const long long row = blockIdx.x;
  const float* s = sc + row * 1024;
  bf16t* p = pr + row * 1024;
  const int tid = threadIdx.x;
  float v[4]; float mx = -1e30f;
  #pragma unroll
  for (int i = 0; i < 4; ++i) { v[i] = s[tid + 256 * i]; mx = fmaxf(mx, v[i]); }
  sred[tid] = mx; __syncthreads();
  for (int st = 128; st > 0; st >>= 1) { if (tid < st) sred[tid] = fmaxf(sred[tid], sred[tid + st]); __syncthreads(); }
  mx = sred[0]; __syncthreads();
  float sum = 0.f;
  #pragma unroll
  for (int i = 0; i < 4; ++i) { v[i] = __expf(v[i] - mx); sum += v[i]; }
  sred[tid] = sum; __syncthreads();
  for (int st = 128; st > 0; st >>= 1) { if (tid < st) sred[tid] += sred[tid + st]; __syncthreads(); }
  const float inv = 1.f / sred[0];
  #pragma unroll
  for (int i = 0; i < 4; ++i) p[tid + 256 * i] = f2bf(v[i] * inv);
}

// depthwise 3-tap conv + SiLU -> bf16
__global__ __launch_bounds__(256)
void k_conv(const float* __restrict__ xn, const float* __restrict__ dww,
            const float* __restrict__ dwb, bf16t* __restrict__ act) {
  const long long id = (long long)blockIdx.x * 256 + threadIdx.x;
  if (id >= (long long)TT * DMD) return;
  const int cidx = (int)(id % DMD);
  const long long t = id / DMD;
  const int l = (int)(t % LL);
  const float left  = (l > 0)      ? xn[id - DMD] : 0.f;
  const float right = (l < LL - 1) ? xn[id + DMD] : 0.f;
  const float dw = left * dww[cidx * 3 + 0] + xn[id] * dww[cidx * 3 + 1] +
                   right * dww[cidx * 3 + 2] + dwb[cidx];
  act[id] = f2bf(siluf(dw));
}

// mix branches, x2, rms2 + moe gate softmax + top-2 (one block = one token)
__global__ __launch_bounds__(256)
void k_mix(const float* __restrict__ res, const float* __restrict__ ssd,
           const float* __restrict__ attn, const float* __restrict__ conv,
           const float* __restrict__ mem, const float* __restrict__ wr,
           const float* __restrict__ n2w, const float* __restrict__ mgw,
           float* __restrict__ x2, bf16t* __restrict__ x2b,
           float* __restrict__ outb, float* __restrict__ wfull) {
  __shared__ float sred[256];
  __shared__ float sg[4];
  const int t = blockIdx.x, tid = threadIdx.x;
  const long long base = (long long)t * DMD;
  const float w0 = wr[t * 4 + 0], w1 = wr[t * 4 + 1], w2 = wr[t * 4 + 2], w3 = wr[t * 4 + 3];
  float xv[4]; float ss = 0.f;
  #pragma unroll
  for (int i = 0; i < 4; ++i) {
    const int cidx = tid * 4 + i;
    const float v = res[base + cidx] + w0 * ssd[base + cidx] + w1 * attn[base + cidx] +
                    w2 * conv[base + cidx] + w3 * mem[base + cidx];
    xv[i] = v; ss += v * v;
    x2[base + cidx] = v; x2b[base + cidx] = f2bf(v); outb[base + cidx] = v;
  }
  sred[tid] = ss; __syncthreads();
  for (int s = 128; s > 0; s >>= 1) { if (tid < s) sred[tid] += sred[tid + s]; __syncthreads(); }
  const float rs = rsqrtf(sred[0] / (float)DMD + 1e-6f);
  __syncthreads();
  if (tid < 4) sg[tid] = 0.f;
  __syncthreads();
  #pragma unroll
  for (int e = 0; e < 4; ++e) {
    float pp = 0.f;
    #pragma unroll
    for (int i = 0; i < 4; ++i) {
      const int cidx = tid * 4 + i;
      pp += xv[i] * rs * n2w[cidx] * mgw[cidx * 4 + e];
    }
    atomicAdd(&sg[e], pp);
  }
  __syncthreads();
  if (tid == 0) {
    float mx = fmaxf(fmaxf(sg[0], sg[1]), fmaxf(sg[2], sg[3]));
    float p[4]; float Z = 0.f;
    #pragma unroll
    for (int e = 0; e < 4; ++e) { p[e] = __expf(sg[e] - mx); Z += p[e]; }
    #pragma unroll
    for (int e = 0; e < 4; ++e) p[e] /= Z;
    int i1 = 0;
    for (int e = 1; e < 4; ++e) if (p[e] > p[i1]) i1 = e;
    int i2 = -1;
    for (int e = 0; e < 4; ++e) { if (e == i1) continue; if (i2 < 0 || p[e] > p[i2]) i2 = e; }
    #pragma unroll
    for (int e = 0; e < 4; ++e) wfull[t * 4 + e] = (e == i1 || e == i2) ? p[e] : 0.f;
  }
}

// GLU: silu(a)*g -> bf16
__global__ __launch_bounds__(256)
void k_glu(const float* __restrict__ h1, bf16t* __restrict__ hact) {
  const long long id = (long long)blockIdx.x * 256 + threadIdx.x;
  if (id >= (long long)TT * FFD) return;
  const long long t = id / FFD; const int f = (int)(id % FFD);
  const float a = h1[t * (2 * FFD) + f];
  const float g = h1[t * (2 * FFD) + FFD + f];
  hact[id] = f2bf(siluf(a) * g);
}

// =====================================================================
// host launch
// =====================================================================
static void gemm(hipStream_t st, const bf16t* A, long long lda,
                 const bf16t* Bmat, long long ldb, float* C, long long ldc,
                 int M, int N, int K,
                 float alpha = 1.f, const float* bias = nullptr, int act = 0,
                 const float* rowscale = nullptr, int rsStride = 1, int accumulate = 0,
                 int batch = 1, int hdiv = 1,
                 long long aO = 0, long long aI = 0, long long bO = 0, long long bI = 0,
                 long long cO = 0, long long cI = 0, int bDiv = 1) {
  GemmP p;
  p.A = A; p.Bm = Bmat; p.C = C; p.lda = lda; p.ldb = ldb; p.ldc = ldc;
  p.M = M; p.N = N; p.K = K; p.batch = batch; p.hdiv = hdiv; p.bDiv = bDiv;
  p.aO = aO; p.aI = aI; p.bO = bO; p.bI = bI; p.cO = cO; p.cI = cI;
  p.alpha = alpha; p.bias = bias; p.rowscale = rowscale; p.rsStride = rsStride;
  p.accumulate = accumulate; p.act = act;
  if ((N % 128) == 0) {
    // wave 64x32, block 128x128
    dim3 g(N / 128, M / 128, batch);
    gemm_bf16<2, 4, 4, 2><<<g, 256, 0, st>>>(p);
  } else {
    // wave 32x32, block 128x64 (attention PV, N=64)
    dim3 g(N / 64, M / 128, batch);
    gemm_bf16<4, 2, 2, 2><<<g, 256, 0, st>>>(p);
  }
}

static void cvt(hipStream_t st, const float* s, bf16t* d, long long n) {
  long long blocks = (n + 255) / 256;
  k_f2b<<<(unsigned)blocks, 256, 0, st>>>(s, d, n);
}
static void cvtT(hipStream_t st, const float* s, bf16t* d, int K, int N, int batch) {
  long long blocks = ((long long)K * N + 255) / 256;
  dim3 g((unsigned)blocks, batch);
  k_f2b_t<<<g, 256, 0, st>>>(s, d, K, N);
}

extern "C" void kernel_launch(void* const* d_in, const int* in_sizes, int n_in,
                              void* d_out, int out_size, void* d_ws, size_t ws_size,
                              hipStream_t stream) {
  const float* x        = (const float*)d_in[0];
  const float* goal     = (const float*)d_in[1];
  const float* mem_out  = (const float*)d_in[2];
  const float* norm1_w  = (const float*)d_in[3];
  const float* norm2_w  = (const float*)d_in[4];
  const float* router_w = (const float*)d_in[5];
  const float* xproj_w  = (const float*)d_in[6];
  const float* Aw       = (const float*)d_in[7];
  const float* Dp       = (const float*)d_in[8];
  const float* gate_w   = (const float*)d_in[9];
  const float* gate_b   = (const float*)d_in[10];
  const float* ssd_o_w  = (const float*)d_in[11];
  const float* q_w      = (const float*)d_in[12];
  const float* k_w      = (const float*)d_in[13];
  const float* v_w      = (const float*)d_in[14];
  const float* o_w      = (const float*)d_in[15];
  const float* qn_w     = (const float*)d_in[16];
  const float* kn_w     = (const float*)d_in[17];
  const float* dw_w     = (const float*)d_in[18];
  const float* dw_b     = (const float*)d_in[19];
  const float* pw_w     = (const float*)d_in[20];
  const float* pw_b     = (const float*)d_in[21];
  const float* moe_gate = (const float*)d_in[22];
  const float* e_w1     = (const float*)d_in[23];
  const float* e_w2     = (const float*)d_in[24];
  const float* e_lin_w  = (const float*)d_in[25];
  const float* e_lin_b  = (const float*)d_in[26];
  float* out = (float*)d_out;

  // ---- workspace carve ----
  char* w = (char*)d_ws;
  size_t cur = 0;
  auto alloc = [&](size_t bytes) -> void* {
    void* p = w + cur;
    cur += (bytes + 255) & ~(size_t)255;
    return p;
  };
  const long long TD = (long long)TT * DMD;

  // all weight copies are bf16 AND transposed to N-major (N x K)
  bf16t* wb_gate = (bf16t*)alloc(DMD * DMD * 2);
  bf16t* wb_ssd  = (bf16t*)alloc(DMD * DMD * 2);
  bf16t* wb_q    = (bf16t*)alloc(DMD * DMD * 2);
  bf16t* wb_k    = (bf16t*)alloc(DMD * 256 * 2);
  bf16t* wb_v    = (bf16t*)alloc(DMD * 256 * 2);
  bf16t* wb_o    = (bf16t*)alloc(DMD * DMD * 2);
  bf16t* wb_pw   = (bf16t*)alloc(DMD * DMD * 2);
  bf16t* wb_w1   = (bf16t*)alloc((size_t)NEXP * DMD * 2 * FFD * 2);
  bf16t* wb_w2   = (bf16t*)alloc((size_t)NEXP * FFD * DMD * 2);
  bf16t* wb_lin  = (bf16t*)alloc((size_t)NEXP * DMD * DMD * 2);

  float* xn      = (float*)alloc(TD * 4);
  bf16t* xnb     = (bf16t*)alloc(TD * 2);
  float* wrout   = (float*)alloc(TT * 4 * 4);
  float* sp      = (float*)alloc((long long)TT * 144 * 4);
  float* states  = (float*)alloc((long long)TT * NH * NST * 4);
  float* yh      = (float*)alloc((long long)TT * NH * 4);
  float* gatebuf = (float*)alloc(TD * 4);
  bf16t* t1b     = (bf16t*)alloc(TD * 2);
  float* qbuf    = (float*)alloc(TD * 4);
  float* kbuf    = (float*)alloc((long long)TT * 256 * 4);
  float* vbuf    = (float*)alloc((long long)TT * 256 * 4);
  bf16t* qbf     = (bf16t*)alloc(TD * 2);
  bf16t* kbf     = (bf16t*)alloc((long long)TT * 256 * 2);
  bf16t* vbf     = (bf16t*)alloc((long long)TT * 256 * 2);
  float* scores  = (float*)alloc((long long)BB * NH * LL * LL * 4);   // 128MB, reused by MoE
  bf16t* prob    = (bf16t*)alloc((long long)BB * NH * LL * LL * 2);
  float* attnf   = (float*)alloc(TD * 4);
  bf16t* attnb   = (bf16t*)alloc(TD * 2);
  float* out_ssd = (float*)alloc(TD * 4);
  float* out_att = (float*)alloc(TD * 4);
  float* out_cnv = (float*)alloc(TD * 4);
  bf16t* convact = (bf16t*)alloc(TD * 2);
  float* x2      = (float*)alloc(TD * 4);
  bf16t* x2b     = (bf16t*)alloc(TD * 2);
  float* wfull   = (float*)alloc(TT * 4 * 4);
  (void)ws_size; (void)n_in; (void)in_sizes; (void)out_size;

  // MoE scratch aliases the (by-then consumed) scores buffer (128MB >= 92MB)
  float* h1   = scores;                                              // 64MB
  bf16t* hact = (bf16t*)((char*)scores + (size_t)64 * 1024 * 1024);  // 16MB
  float* h2   = (float*)((char*)scores + (size_t)80 * 1024 * 1024);  // 8MB
  bf16t* h2b  = (bf16t*)((char*)scores + (size_t)88 * 1024 * 1024);  // 4MB

  // ---- 1. weight conversions f32 -> bf16, transposed to (N x K) ----
  cvtT(stream, gate_w, wb_gate, DMD, DMD, 1);
  cvtT(stream, ssd_o_w, wb_ssd, DMD, DMD, 1);
  cvtT(stream, q_w, wb_q, DMD, DMD, 1);
  cvtT(stream, k_w, wb_k, DMD, 256, 1);
  cvtT(stream, v_w, wb_v, DMD, 256, 1);
  cvtT(stream, o_w, wb_o, DMD, DMD, 1);
  cvtT(stream, pw_w, wb_pw, DMD, DMD, 1);
  cvtT(stream, e_w1, wb_w1, DMD, 2 * FFD, NEXP);
  cvtT(stream, e_w2, wb_w2, FFD, DMD, NEXP);
  cvtT(stream, e_lin_w, wb_lin, DMD, DMD, NEXP);

  // ---- 2. norm1 + router, ssm projection ----
  k_rms1<<<TT, 256, 0, stream>>>(x, goal, norm1_w, router_w, xn, xnb, wrout);
  k_ssmproj<<<TT, 256, 0, stream>>>(xn, xproj_w, sp);

  // ---- 3. SSD branch ----
  gemm(stream, xnb, DMD, wb_gate, DMD, gatebuf, DMD, TT, DMD, DMD,
       1.f, gate_b, /*act=*/1);
  k_scan<<<(BB * NH * NST + 255) / 256, 256, 0, stream>>>(sp, Aw, states);
  k_yh<<<(TT * NH + 255) / 256, 256, 0, stream>>>(states, sp, yh);
  k_t1<<<(unsigned)((TD + 255) / 256), 256, 0, stream>>>(yh, xn, Dp, gatebuf, t1b);
  gemm(stream, t1b, DMD, wb_ssd, DMD, out_ssd, DMD, TT, DMD, DMD);

  // ---- 4. attention branch ----
  gemm(stream, xnb, DMD, wb_q, DMD, qbuf, DMD, TT, DMD, DMD);
  gemm(stream, xnb, DMD, wb_k, DMD, kbuf, 256, TT, 256, DMD);
  gemm(stream, xnb, DMD, wb_v, DMD, vbuf, 256, TT, 256, DMD);
  k_rope<<<BB * NH * LL, 64, 0, stream>>>(qbuf, qn_w, qbf, NH);
  k_rope<<<BB * NKV * LL, 64, 0, stream>>>(kbuf, kn_w, kbf, NKV);
  k_vperm<<<(BB * NKV * LL * HDI + 255) / 256, 256, 0, stream>>>(vbuf, vbf);

  const long long LHD = (long long)LL * HDI;
  // scores = 0.125 * Q K^T  (K rows are naturally N-major: (key_l, hd))
  gemm(stream, qbf, HDI, kbf, HDI, scores, LL, LL, LL, HDI,
       0.125f, nullptr, 0, nullptr, 1, 0,
       /*batch=*/BB * NH, /*hdiv=*/NH,
       /*aO=*/(long long)NH * LHD, /*aI=*/LHD,
       /*bO=*/(long long)NKV * LHD, /*bI=*/LHD,
       /*cO=*/(long long)NH * LL * LL, /*cI=*/(long long)LL * LL,
       /*bDiv=*/NH / NKV);
  k_softmax<<<BB * NH * LL, 256, 0, stream>>>(scores, prob);
  // attn = P V   (V transposed to (hd, l) N-major); write into (b, l, h*64+hd)
  gemm(stream, prob, LL, vbf, LL, attnf, DMD, LL, HDI, LL,
       1.f, nullptr, 0, nullptr, 1, 0,
       /*batch=*/BB * NH, /*hdiv=*/NH,
       /*aO=*/(long long)NH * LL * LL, /*aI=*/(long long)LL * LL,
       /*bO=*/(long long)NKV * LHD, /*bI=*/LHD,
       /*cO=*/(long long)LL * DMD, /*cI=*/HDI,
       /*bDiv=*/NH / NKV);
  cvt(stream, attnf, attnb, TD);
  gemm(stream, attnb, DMD, wb_o, DMD, out_att, DMD, TT, DMD, DMD);

  // ---- 5. conv branch ----
  k_conv<<<(unsigned)((TD + 255) / 256), 256, 0, stream>>>(xn, dw_w, dw_b, convact);
  gemm(stream, convact, DMD, wb_pw, DMD, out_cnv, DMD, TT, DMD, DMD,
       1.f, pw_b, 0);

  // ---- 6. mix + norm2 + moe gating; out = x2 ----
  k_mix<<<TT, 256, 0, stream>>>(x, out_ssd, out_att, out_cnv, mem_out, wrout,
                                norm2_w, moe_gate, x2, x2b, out, wfull);

  // ---- 7. dense MoE, accumulated into out ----
  for (int e = 0; e < NEXP; ++e) {
    const bf16t* w1 = wb_w1 + (size_t)e * DMD * 2 * FFD;
    const bf16t* w2 = wb_w2 + (size_t)e * FFD * DMD;
    const bf16t* wl = wb_lin + (size_t)e * DMD * DMD;
    gemm(stream, x2b, DMD, w1, DMD, h1, 2 * FFD, TT, 2 * FFD, DMD);
    k_glu<<<(unsigned)(((long long)TT * FFD + 255) / 256), 256, 0, stream>>>(h1, hact);
    gemm(stream, hact, FFD, w2, FFD, h2, DMD, TT, DMD, FFD);
    cvt(stream, h2, h2b, TD);
    gemm(stream, h2b, DMD, wl, DMD, out, DMD, TT, DMD, DMD,
         1.f, e_lin_b + (size_t)e * DMD, /*act=*/0,
         /*rowscale=*/wfull + e, /*rsStride=*/4, /*accumulate=*/1);
  }
}
